// CausalAttention_62397284876912
// MI455X (gfx1250) — matrix operand order repaired and checked by reading, not compiled, not run on hardware
//
#include <hip/hip_runtime.h>
#include <hip/hip_bf16.h>

typedef _Float16 f16;
typedef __attribute__((ext_vector_type(16))) _Float16 v16h;
typedef __attribute__((ext_vector_type(8)))  _Float16 v8h;
typedef __attribute__((ext_vector_type(4)))  _Float16 v4h;
typedef __attribute__((ext_vector_type(8)))  float    v8f;
typedef __attribute__((ext_vector_type(4)))  float    f4;

#define NSEQ 2048
#define EMB  512
#define NH   8
#define HD   64
#define BATCH 4

static __device__ inline v8f wmma_f16(v16h a, v16h b, v8f c) {
    return __builtin_amdgcn_wmma_f32_16x16x32_f16(false, a, false, b, (short)0, c, false, false);
}

// A-fragment (16x32) from f16 row: runs [kk+half*8,+8) and [kk+16+half*8,+8).
static __device__ inline v16h load_a_frag_f16(const f16* row, int kk, int half) {
    v8h lo = *(const v8h*)(row + kk + (half << 3));
    v8h hi = *(const v8h*)(row + kk + 16 + (half << 3));
    v16h a;
#pragma unroll
    for (int i = 0; i < 8; ++i) { a[i] = lo[i]; a[8 + i] = hi[i]; }
    return a;
}

// B-fragment (32x16) from f16 row: runs [kk+half*8,+8) and [kk+16+half*8,+8).
static __device__ inline v16h load_b_frag_f16(const f16* row, int kk, int half) {
    v8h lo = *(const v8h*)(row + kk + (half << 3));
    v8h hi = *(const v8h*)(row + kk + 16 + (half << 3));
    v16h b;
#pragma unroll
    for (int i = 0; i < 8; ++i) { b[i] = lo[i]; b[8 + i] = hi[i]; }
    return b;
}

// A-fragment from f32 (LDS) row: 4x b128 + cvt (used for w @ V only).
static __device__ inline v16h load_a_frag_f32(const float* row, int kk, int half) {
    f4 t0 = *(const f4*)(row + kk + (half << 3));
    f4 t1 = *(const f4*)(row + kk + (half << 3) + 4);
    f4 t2 = *(const f4*)(row + kk + 16 + (half << 3));
    f4 t3 = *(const f4*)(row + kk + 16 + (half << 3) + 4);
    v16h a;
#pragma unroll
    for (int i = 0; i < 4; ++i) {
        a[i]      = (f16)t0[i];
        a[4 + i]  = (f16)t1[i];
        a[8 + i]  = (f16)t2[i];
        a[12 + i] = (f16)t3[i];
    }
    return a;
}

// ---------------------------------------------------------------------------
// f32 -> f16 pack (elementwise), 4 elems per thread.
// ---------------------------------------------------------------------------
__global__ __launch_bounds__(256) void cvt_f16_kernel(
    const float* __restrict__ src, f16* __restrict__ dst, int n4)
{
    const int i = blockIdx.x * 256 + threadIdx.x;
    if (i >= n4) return;
    const f4 v = ((const f4*)src)[i];
    v4h o;
#pragma unroll
    for (int q = 0; q < 4; ++q) o[q] = (f16)v[q];
    ((v4h*)dst)[i] = o;
}

// ---------------------------------------------------------------------------
// GEMM: Y[m][n] = sum_k X[m][k] * W[n][k] + bias[n]   (X, W already f16)
// Each wave: 16x64 strip (4 n-tiles share one A fragment).
// Ping-pong double buffering (k unrolled by 2) -> no register-copy swaps.
// mode 0: f16 out [B,H,N,D].  mode 1: f32 out [M,512].  mode 2: f16 out [B,H,D,N].
// ---------------------------------------------------------------------------
__global__ __launch_bounds__(128) void gemm_f16_kernel(
    const f16* __restrict__ X, const f16* __restrict__ W,
    const float* __restrict__ bias, void* __restrict__ dst, int mode)
{
    const int wave = threadIdx.x >> 5;
    const int lane = threadIdx.x & 31;
    const int half = lane >> 4;
    const int lm   = lane & 15;

    const int task = blockIdx.x * 4 + wave;     // 0..4095
    const int mt   = task >> 3;                 // 512 m-tiles
    const int ng   = task & 7;                  // n-group: 4 tiles of 16

    const f16* xrow = X + (size_t)(mt * 16 + lm) * EMB;
    const f16* wrow[4];
#pragma unroll
    for (int t = 0; t < 4; ++t)
        wrow[t] = W + (size_t)(ng * 64 + t * 16 + lm) * EMB;

    v8f acc[4] = {{}, {}, {}, {}};

    v16h aA = load_a_frag_f16(xrow, 0, half);
    v16h bA[4];
#pragma unroll
    for (int t = 0; t < 4; ++t) bA[t] = load_b_frag_f16(wrow[t], 0, half);

    for (int kk = 0; kk < EMB; kk += 64) {       // 8 iterations, 2 chunks each
        const int k1 = kk + 32;                  // always < EMB
        const v16h aB = load_a_frag_f16(xrow, k1, half);
        v16h bB[4];
#pragma unroll
        for (int t = 0; t < 4; ++t) bB[t] = load_b_frag_f16(wrow[t], k1, half);
#pragma unroll
        for (int t = 0; t < 4; ++t) acc[t] = wmma_f16(aA, bA[t], acc[t]);

        const int k2 = (kk + 64 < EMB) ? kk + 64 : k1;   // clamped redundant prefetch
        aA = load_a_frag_f16(xrow, k2, half);
#pragma unroll
        for (int t = 0; t < 4; ++t) bA[t] = load_b_frag_f16(wrow[t], k2, half);
#pragma unroll
        for (int t = 0; t < 4; ++t) acc[t] = wmma_f16(aB, bB[t], acc[t]);
    }

#pragma unroll
    for (int t = 0; t < 4; ++t) {
        const int n  = ng * 64 + t * 16 + lm;
        const float bn = bias[n];
#pragma unroll
        for (int v = 0; v < 8; ++v) {
            const int m = mt * 16 + v + (half << 3);
            const float y = acc[t][v] + bn;
            if (mode == 0) {
                const int bb = m >> 11, i = m & (NSEQ - 1);
                const int hh = n >> 6,  d = n & (HD - 1);
                ((f16*)dst)[(((size_t)(bb * NH + hh)) * NSEQ + i) * HD + d] = (f16)y;
            } else if (mode == 1) {
                ((float*)dst)[(size_t)m * EMB + n] = y;
            } else { // transposed head layout [B,H,D,N]
                const int bb = m >> 11, i = m & (NSEQ - 1);
                const int hh = n >> 6,  d = n & (HD - 1);
                ((f16*)dst)[(((size_t)(bb * NH + hh)) * HD + d) * NSEQ + i] = (f16)y;
            }
        }
    }
}

// ---------------------------------------------------------------------------
// Fused attention: one workgroup per (batch, 16-row block). Loops all 8 heads.
// LDS: 16x2048 score strip + 16x2048 head-average strip (~263 KB of 320 KB).
// Q,K: [B,H,N,D] f16.  Vt: [B,H,D,N] f16.  O written as f16 [B*N, E].
// ---------------------------------------------------------------------------
__global__ __launch_bounds__(128) void attn_kernel(
    const f16* __restrict__ Q, const f16* __restrict__ K,
    const f16* __restrict__ Vt, const float* __restrict__ energy,
    f16* __restrict__ O, float* __restrict__ avg_out)
{
    __shared__ __align__(16) float S[16][NSEQ];     // scores -> p -> w
    __shared__ __align__(16) float AVG[16][NSEQ];   // head-averaged attention
    __shared__ float ti[16], mrow[16], lrow[16], denomr[16];
    __shared__ float red[16][8];

    const int wg = blockIdx.x;        // 0..511
    const int b  = wg >> 7;
    const int i0 = (wg & 127) * 16;

    const int tid  = threadIdx.x;
    const int wave = tid >> 5;
    const int lane = tid & 31;
    const int half = lane >> 4;
    const int lm   = lane & 15;

    if (tid < 16) ti[tid] = tanhf(energy[b * NSEQ + i0 + tid]);
    __syncthreads();

    const float scale = 0.125f;       // 1/sqrt(64)
    const int sr = tid >> 3;          // scan row 0..15
    const int c0 = (tid & 7) * 256;   // scan col chunk

    for (int h = 0; h < NH; ++h) {
        const f16* Qh  = Q  + ((size_t)(b * NH + h)) * NSEQ * HD;
        const f16* Kh  = K  + ((size_t)(b * NH + h)) * NSEQ * HD;
        const f16* Vth = Vt + ((size_t)(b * NH + h)) * HD * NSEQ;

        // --- Q A-fragments (k-chunks 0 and 32), row m = i0 + lm ---
        const f16* qrow = Qh + (size_t)(i0 + lm) * HD;
        const v16h aq0 = load_a_frag_f16(qrow, 0, half);
        const v16h aq1 = load_a_frag_f16(qrow, 32, half);

        // --- scores: S = Q K^T * scale + 2*tanh(e_j - e_i), ping-pong x2 ---
        {
            v16h b0A = load_b_frag_f16(Kh + (size_t)(wave * 16 + lm) * HD, 0, half);
            v16h b1A = load_b_frag_f16(Kh + (size_t)(wave * 16 + lm) * HD, 32, half);
            float ejA = energy[b * NSEQ + wave * 16 + lm];
            for (int jt = wave; jt < NSEQ / 16; jt += 8) {   // pairs (jt, jt+4)
                const int j1 = jt + 4;                       // always valid
                const f16* kr1 = Kh + (size_t)(j1 * 16 + lm) * HD;
                const v16h b0B = load_b_frag_f16(kr1, 0, half);
                const v16h b1B = load_b_frag_f16(kr1, 32, half);
                const float ejB = energy[b * NSEQ + j1 * 16 + lm];

                {   // tile jt (A buffers)
                    v8f c = {};
                    c = wmma_f16(aq0, b0A, c);
                    c = wmma_f16(aq1, b1A, c);
                    const float tj = tanhf(ejA);
                    const int j0 = jt * 16;
#pragma unroll
                    for (int v = 0; v < 8; ++v) {
                        const int row = v + (half << 3);
                        const float tir = ti[row];
                        S[row][j0 + lm] = c[v] * scale
                                        + 2.0f * (tj - tir) / (1.0f - tj * tir);
                    }
                }

                const int j2 = (jt + 8 < NSEQ / 16) ? jt + 8 : j1;  // clamped prefetch
                const f16* kr2 = Kh + (size_t)(j2 * 16 + lm) * HD;
                b0A = load_b_frag_f16(kr2, 0, half);
                b1A = load_b_frag_f16(kr2, 32, half);
                ejA = energy[b * NSEQ + j2 * 16 + lm];

                {   // tile j1 (B buffers)
                    v8f c = {};
                    c = wmma_f16(aq0, b0B, c);
                    c = wmma_f16(aq1, b1B, c);
                    const float tj = tanhf(ejB);
                    const int j0 = j1 * 16;
#pragma unroll
                    for (int v = 0; v < 8; ++v) {
                        const int row = v + (half << 3);
                        const float tir = ti[row];
                        S[row][j0 + lm] = c[v] * scale
                                        + 2.0f * (tj - tir) / (1.0f - tj * tir);
                    }
                }
            }
        }
        __syncthreads();

        // --- row max ---
        float mx = -3.4e38f;
        for (int c = 0; c < 256; ++c) mx = fmaxf(mx, S[sr][c0 + c]);
        red[sr][tid & 7] = mx;
        __syncthreads();
        if (tid < 16) {
            float m = red[tid][0];
            for (int q = 1; q < 8; ++q) m = fmaxf(m, red[tid][q]);
            mrow[tid] = m;
        }
        __syncthreads();

        // --- p = exp(s-m), row sum l ---
        {
            const float m = mrow[sr];
            float lsum = 0.f;
            for (int c = 0; c < 256; ++c) {
                const float p = expf(S[sr][c0 + c] - m);
                S[sr][c0 + c] = p;
                lsum += p;
            }
            red[sr][tid & 7] = lsum;
        }
        __syncthreads();
        if (tid < 16) {
            float s = 0.f;
            for (int q = 0; q < 8; ++q) s += red[tid][q];
            lrow[tid] = s;
        }
        __syncthreads();

        // --- gated sum G = sum p * [p/l > 0.01] ---
        {
            const float thr = 0.01f * lrow[sr];
            float g = 0.f;
            for (int c = 0; c < 256; ++c) {
                const float p = S[sr][c0 + c];
                g += (p > thr) ? p : 0.f;
            }
            red[sr][tid & 7] = g;
        }
        __syncthreads();
        if (tid < 16) {
            float s = 0.f;
            for (int q = 0; q < 8; ++q) s += red[tid][q];
            denomr[tid] = fmaxf(s, 1e-9f * lrow[tid]);
        }
        __syncthreads();

        // --- finalize w, accumulate head average ---
        {
            const float thr = 0.01f * lrow[sr];
            const float dn  = denomr[sr];
            for (int c = 0; c < 256; ++c) {
                const float p = S[sr][c0 + c];
                const float w = (p > thr) ? (p / dn) : 0.f;
                S[sr][c0 + c] = w;
                const float prev = (h == 0) ? 0.f : AVG[sr][c0 + c];
                AVG[sr][c0 + c] = prev + w * 0.125f;
            }
        }
        __syncthreads();

        // --- context = w @ V : wave computes d-tile [wave*16,+16), ping-pong x2 ---
        {
            const int dt = wave;
            const float* srow = &S[lm][0];
            const f16* vrow = Vth + (size_t)(dt * 16 + lm) * NSEQ;  // B col n = d
            v8f acc = {};
            v16h bvA = load_b_frag_f16(vrow, 0, half);
            for (int j0 = 0; j0 < NSEQ; j0 += 64) {   // pairs (j0, j0+32)
                const int j1 = j0 + 32;               // always valid
                const v16h bvB = load_b_frag_f16(vrow, j1, half);
                const v16h ap0 = load_a_frag_f32(srow, j0, half);
                acc = wmma_f16(ap0, bvA, acc);

                const int j2 = (j0 + 64 < NSEQ) ? j0 + 64 : j1;  // clamped prefetch
                bvA = load_b_frag_f16(vrow, j2, half);
                const v16h ap1 = load_a_frag_f32(srow, j1, half);
                acc = wmma_f16(ap1, bvB, acc);
            }
#pragma unroll
            for (int v = 0; v < 8; ++v) {
                const int row = v + (half << 3);
                O[((size_t)b * NSEQ + i0 + row) * EMB + h * HD + dt * 16 + lm] = (f16)acc[v];
            }
        }
        __syncthreads();
    }

    // --- write attn_avg strip: 16-byte nontemporal streaming stores ---
    for (int idx = tid; idx < 16 * (NSEQ / 4); idx += 128) {
        const int r  = idx >> 9;            // NSEQ/4 = 512 vectors per row
        const int c4 = idx & 511;
        const f4 val = *(const f4*)&AVG[r][c4 * 4];
        f4* dstp = (f4*)(avg_out + ((size_t)b * NSEQ + i0 + r) * NSEQ) + c4;
        __builtin_nontemporal_store(val, dstp);
    }
}

// ---------------------------------------------------------------------------
extern "C" void kernel_launch(void* const* d_in, const int* in_sizes, int n_in,
                              void* d_out, int out_size, void* d_ws, size_t ws_size,
                              hipStream_t stream) {
    const float* query  = (const float*)d_in[0];
    const float* key_   = (const float*)d_in[1];
    const float* value  = (const float*)d_in[2];
    const float* energy = (const float*)d_in[3];
    const float* Wq = (const float*)d_in[4];  const float* bq = (const float*)d_in[5];
    const float* Wk = (const float*)d_in[6];  const float* bk = (const float*)d_in[7];
    const float* Wv = (const float*)d_in[8];  const float* bv = (const float*)d_in[9];
    const float* Wo = (const float*)d_in[10]; const float* bo = (const float*)d_in[11];

    float* out = (float*)d_out;                                // [B, N, E]
    float* avg = out + (size_t)BATCH * NSEQ * EMB;             // [B, N, N]

    // workspace layout (f16 unless noted)
    char* ws = (char*)d_ws;
    const size_t actBytes = (size_t)BATCH * NSEQ * EMB * sizeof(f16);  // 8 MB
    const size_t wBytes   = (size_t)EMB * EMB * sizeof(f16);           // 512 KB
    f16* Xq = (f16*)(ws);                         // packed inputs
    f16* Xk = (f16*)(ws + actBytes);
    f16* Xv = (f16*)(ws + 2 * actBytes);
    f16* Qh = (f16*)(ws + 3 * actBytes);          // projected, head-split
    f16* Kh = (f16*)(ws + 4 * actBytes);
    f16* Vt = (f16*)(ws + 5 * actBytes);
    f16* Oc = (f16*)(ws + 6 * actBytes);          // context [B*N, E]
    f16* Wqh = (f16*)(ws + 7 * actBytes);
    f16* Wkh = (f16*)(ws + 7 * actBytes + wBytes);
    f16* Wvh = (f16*)(ws + 7 * actBytes + 2 * wBytes);
    f16* Woh = (f16*)(ws + 7 * actBytes + 3 * wBytes);

    const int nAct4 = BATCH * NSEQ * EMB / 4;     // 1,048,576
    const int nW4   = EMB * EMB / 4;              // 65,536
    cvt_f16_kernel<<<(nAct4 + 255) / 256, 256, 0, stream>>>(query, Xq, nAct4);
    cvt_f16_kernel<<<(nAct4 + 255) / 256, 256, 0, stream>>>(key_,  Xk, nAct4);
    cvt_f16_kernel<<<(nAct4 + 255) / 256, 256, 0, stream>>>(value, Xv, nAct4);
    cvt_f16_kernel<<<(nW4 + 255) / 256, 256, 0, stream>>>(Wq, Wqh, nW4);
    cvt_f16_kernel<<<(nW4 + 255) / 256, 256, 0, stream>>>(Wk, Wkh, nW4);
    cvt_f16_kernel<<<(nW4 + 255) / 256, 256, 0, stream>>>(Wv, Wvh, nW4);
    cvt_f16_kernel<<<(nW4 + 255) / 256, 256, 0, stream>>>(Wo, Woh, nW4);

    gemm_f16_kernel<<<1024, 128, 0, stream>>>(Xq, Wqh, bq, (void*)Qh, 0);
    gemm_f16_kernel<<<1024, 128, 0, stream>>>(Xk, Wkh, bk, (void*)Kh, 0);
    gemm_f16_kernel<<<1024, 128, 0, stream>>>(Xv, Wvh, bv, (void*)Vt, 2);

    attn_kernel<<<BATCH * (NSEQ / 16), 128, 0, stream>>>(Qh, Kh, Vt, energy, Oc, avg);

    gemm_f16_kernel<<<1024, 128, 0, stream>>>(Oc, Woh, bo, (void*)out, 1);

    (void)in_sizes; (void)n_in; (void)out_size; (void)ws_size;
}
